// ChnAttn_19636590477367
// MI455X (gfx1250) — compile-verified, matrix operand order checked by reading, not run on hardware
//
#include <hip/hip_runtime.h>

// ---------------------------------------------------------------------------
// Types for CDNA5 WMMA (wave32): D(16x16 f32) = A(16x32 bf16) x B(32x16 bf16) + C
// ---------------------------------------------------------------------------
typedef __attribute__((ext_vector_type(16))) __bf16        v16bf;
typedef __attribute__((ext_vector_type(8)))  float         v8f;
typedef __attribute__((ext_vector_type(8)))  unsigned int  v8u;

union FragU { v8u u; v16bf b; };

// Problem constants (fixed shapes from the reference)
#define BN        8192      // B*H*W*T tokens
#define NMOD      5         // 3 s2 + 2 s1
#define DIM       768
#define ADIM      256
#define NHEADS    8
#define HD        32
#define TOK_PER_WG 16
#define ROWS      (TOK_PER_WG * NMOD)   // 80
#define OUT_ELEMS (BN * DIM)            // 6291456

// Swizzled-weight tile counts
#define WIN_KT   24   // 768/32
#define WIN_NT   16   // 256/16
#define WV_KT    8    // 256/32
#define WV_NT    16
#define WOUT_KT  8    // 256/32
#define WOUT_NT  48   // 768/16
#define WIN_SZ   (WIN_KT*WIN_NT*32*8)    // 98304 uints
#define WV_SZ    (WV_KT*WV_NT*32*8)      // 32768
#define WOUT_SZ  (WOUT_KT*WOUT_NT*32*8)  // 98304

// LDS partition (bytes)
#define SA_OFF    0                       // 80 x 136 bf16 (K chunk of 128 + pad)
#define SX_OFF    21760                   // 80 x 264 bf16
#define SV_OFF    64000                   // 80 x 260 f32
#define SQK_OFF   147200                  // 8 x 256 f32
#define SQB_OFF   155392                  // 8 f32
#define SATTN_OFF 155424                  // 16 x 8 x 5 f32
#define SHAS_OFF  157984                  // 16 int
#define SY_OFF    158048                  // 16 x 264 bf16
#define SMEM_SZ   166496

// ---------------------------------------------------------------------------
// bf16 helpers (manual round-to-nearest-even)
// ---------------------------------------------------------------------------
__device__ __forceinline__ unsigned f2bf1(float x) {
    unsigned u = __float_as_uint(x);
    return (u + 0x7FFFu + ((u >> 16) & 1u)) >> 16;
}
__device__ __forceinline__ unsigned pack2(float lo, float hi) {
    return f2bf1(lo) | (f2bf1(hi) << 16);
}
__device__ __forceinline__ float bf2f(unsigned short s) {
    return __uint_as_float(((unsigned)s) << 16);
}

// A-fragment (16x32 bf16, M on lanes, ISA 7.12.2 16-bit A layout) from LDS
__device__ __forceinline__ v16bf load_a(const unsigned short* base, int row0,
                                        int stride, int k0, int m, int half) {
    const unsigned short* p = base + (row0 + m) * stride + k0 + half * 8;
    FragU f;
#pragma unroll
    for (int j = 0; j < 8; ++j) {
        int kk = ((j < 4) ? 0 : 16) + 2 * (j & 3);
        f.u[j] = *(const unsigned int*)(p + kk);
    }
    return f.b;
}

// B-fragment from pre-swizzled global weights: 32 contiguous bytes per lane
__device__ __forceinline__ v16bf load_b(const unsigned int* p) {
    FragU f;
    f.u = *(const v8u*)p;
    return f.b;
}

// ---------------------------------------------------------------------------
// Prep 1: fold the learned query into W_k:  qk[h][a] = scale * q[h,:] . W_k[h*32:,a]
// ---------------------------------------------------------------------------
__global__ void prep_qk(const float* __restrict__ q, const float* __restrict__ Wk,
                        const float* __restrict__ bk,
                        float* __restrict__ qk, float* __restrict__ qb) {
    const float scale = 0.17677669529663687f; // 32^-0.5
    int tid = threadIdx.x;
    for (int o = tid; o < NHEADS * ADIM; o += 256) {
        int h = o >> 8, a = o & 255;
        float s = 0.f;
#pragma unroll 8
        for (int d = 0; d < HD; ++d)
            s += q[h * HD + d] * Wk[(h * HD + d) * ADIM + a];
        qk[o] = s * scale;
    }
    if (tid < NHEADS) {
        float s = 0.f;
        for (int d = 0; d < HD; ++d) s += q[tid * HD + d] * bk[tid * HD + d];
        qb[tid] = s * scale;
    }
}

// ---------------------------------------------------------------------------
// Prep 2: convert W_in / W_v / W_out (f32, row = output feature) to bf16 and
// swizzle into exact WMMA B-fragment order:
//   dst[((kt*NT + nt)*32 + lane)*8 + j] = pack( W[n, kb], W[n, kb+1] )
//   n  = nt*16 + lane%16 ;  kb = kt*32 + (j<4?0:16) + (lane/16)*8 + 2*(j%4)
// ---------------------------------------------------------------------------
__global__ void prep_weights(const float* __restrict__ Win,
                             const float* __restrict__ Wv,
                             const float* __restrict__ Wout,
                             unsigned* __restrict__ winswz,
                             unsigned* __restrict__ wvswz,
                             unsigned* __restrict__ woutswz) {
    const int total = WIN_SZ + WV_SZ + WOUT_SZ;
    for (int u = blockIdx.x * blockDim.x + threadIdx.x; u < total;
         u += gridDim.x * blockDim.x) {
        const float* W;
        unsigned* dst;
        int idx, NT, Kdim;
        if (u < WIN_SZ)               { idx = u;                 W = Win;  dst = winswz;  NT = WIN_NT;  Kdim = DIM;  }
        else if (u < WIN_SZ + WV_SZ)  { idx = u - WIN_SZ;        W = Wv;   dst = wvswz;   NT = WV_NT;   Kdim = ADIM; }
        else                          { idx = u - WIN_SZ - WV_SZ; W = Wout; dst = woutswz; NT = WOUT_NT; Kdim = ADIM; }
        int j = idx & 7;
        int lane = (idx >> 3) & 31;
        int tn = idx >> 8;
        int kt = tn / NT, nt = tn % NT;
        int n = nt * 16 + (lane & 15);
        int kb = kt * 32 + ((j < 4) ? 0 : 16) + (lane >> 4) * 8 + 2 * (j & 3);
        dst[idx] = pack2(W[n * Kdim + kb], W[n * Kdim + kb + 1]);
    }
}

// ---------------------------------------------------------------------------
// Main fused kernel: one workgroup (8 waves) per 16 tokens.
//   GEMM1 : X = cat_tokens @ W_in^T + b_in        (80x768 * 768x256, bf16 WMMA)
//   logits: X . qk  (query folded into W_k) -> masked softmax (VALU)
//   GEMM2 : V = X @ W_v^T + b_v                   (80x256 * 256x256)
//   combine: y[t] = sum_n attn[t,h,n] * v[t,n]    (VALU)
//   GEMM3 : O = Y @ W_out^T + b_out, zero if no encoder token
// ---------------------------------------------------------------------------
__global__ __launch_bounds__(256, 1) void chnattn_main(
    const float* __restrict__ s2, const float* __restrict__ s1,
    const float* __restrict__ b_in, const float* __restrict__ b_v,
    const float* __restrict__ b_out,
    const int* __restrict__ s2m, const int* __restrict__ s1m,
    const unsigned* __restrict__ winswz, const unsigned* __restrict__ wvswz,
    const unsigned* __restrict__ woutswz,
    const float* __restrict__ qk, const float* __restrict__ qb,
    float* __restrict__ out) {
    extern __shared__ __align__(16) char smem[];
    unsigned short* sA   = (unsigned short*)(smem + SA_OFF);
    unsigned short* sX   = (unsigned short*)(smem + SX_OFF);
    float*          sV   = (float*)(smem + SV_OFF);
    float*          sQK  = (float*)(smem + SQK_OFF);
    float*          sQB  = (float*)(smem + SQB_OFF);
    float*          sAt  = (float*)(smem + SATTN_OFF);
    int*            sHas = (int*)(smem + SHAS_OFF);
    unsigned short* sY   = (unsigned short*)(smem + SY_OFF);

    const int tid  = threadIdx.x;
    const int lane = tid & 31;
    const int wave = tid >> 5;
    const int m16  = lane & 15;
    const int half = lane >> 4;
    const long tkbase = (long)blockIdx.x * TOK_PER_WG;
    float* outMask = out + OUT_ELEMS;

    // Stage qk/qb into LDS (reused 640x256 times per WG)
    for (int i = tid; i < NHEADS * ADIM; i += 256) sQK[i] = qk[i];
    if (tid < NHEADS) sQB[tid] = qb[tid];

    // ---------------- Phase 1+2: GEMM1 with K streamed in 128-wide chunks ----
    v8f acc1[10];
#pragma unroll
    for (int i = 0; i < 10; ++i) acc1[i] = (v8f)0.0f;

    for (int chunk = 0; chunk < 6; ++chunk) {
        __syncthreads();
        // stage 80x128 f32 -> bf16 into sA (float4 loads, 2 packed dword stores)
        for (int u = tid; u < ROWS * 32; u += 256) {
            int r = u >> 5, c4 = u & 31;
            int t = r / NMOD, nm = r - t * NMOD;
            long tk = tkbase + t;
            const float* src = (nm < 3) ? (s2 + (tk * 3 + nm) * DIM)
                                        : (s1 + (tk * 2 + (nm - 3)) * DIM);
            const float4 v = *(const float4*)(src + chunk * 128 + c4 * 4);
            unsigned* d = (unsigned*)(sA + r * 136 + c4 * 4);
            d[0] = pack2(v.x, v.y);
            d[1] = pack2(v.z, v.w);
        }
        __syncthreads();
#pragma unroll
        for (int i = 0; i < 10; ++i) {
            int tile = wave + 8 * i;
            int mt = tile >> 4, nt = tile & 15;
#pragma unroll
            for (int kt = 0; kt < 4; ++kt) {
                v16bf a = load_a(sA, mt * 16, 136, kt * 32, m16, half);
                v16bf b = load_b(winswz + (((chunk * 4 + kt) * WIN_NT + nt) * 32 + lane) * 8);
                acc1[i] = __builtin_amdgcn_wmma_f32_16x16x32_bf16(
                    false, a, false, b, (short)0, acc1[i], false, false);
            }
        }
    }
    // Epilogue: X = acc + b_in  -> LDS (bf16)
#pragma unroll
    for (int i = 0; i < 10; ++i) {
        int tile = wave + 8 * i;
        int mt = tile >> 4, nt = tile & 15;
        int n = nt * 16 + m16;
        float bi = b_in[n];
#pragma unroll
        for (int ii = 0; ii < 8; ++ii) {
            int m = mt * 16 + ii + 8 * half;
            sX[m * 264 + n] = (unsigned short)f2bf1(acc1[i][ii] + bi);
        }
    }
    __syncthreads();

    // ---------------- Phase 3: logits (query folded into W_k) + softmax -----
    for (int idx = tid; idx < ROWS * NHEADS; idx += 256) {
        int r = idx >> 3, h = idx & 7;
        int t = r / NMOD, n = r - t * NMOD;
        long tk = tkbase + t;
        int mv = (n < 3) ? s2m[tk * 3 + n] : s1m[tk * 2 + (n - 3)];
        float logit = -1e30f;
        if (mv == 1) {
            const unsigned short* xr = sX + r * 264;
            const float* qh = sQK + h * ADIM;
            float s = 0.f;
#pragma unroll 8
            for (int a = 0; a < ADIM; ++a) s += bf2f(xr[a]) * qh[a];
            logit = s + sQB[h];
        }
        sAt[(t * NHEADS + h) * NMOD + n] = logit;
    }
    if (tid < TOK_PER_WG) {
        long tk = tkbase + tid;
        int any = 0;
        for (int n = 0; n < 3; ++n) any |= (s2m[tk * 3 + n] == 1);
        for (int n = 0; n < 2; ++n) any |= (s1m[tk * 2 + n] == 1);
        sHas[tid] = any;
        outMask[tk] = any ? 1.0f : 0.0f;
    }
    __syncthreads();
    if (tid < TOK_PER_WG * NHEADS) {
        float* L = sAt + tid * NMOD;
        float mx = L[0];
#pragma unroll
        for (int n = 1; n < NMOD; ++n) mx = fmaxf(mx, L[n]);
        float e[NMOD], s = 0.f;
#pragma unroll
        for (int n = 0; n < NMOD; ++n) { e[n] = __expf(L[n] - mx); s += e[n]; }
        float inv = 1.0f / s;
#pragma unroll
        for (int n = 0; n < NMOD; ++n) L[n] = e[n] * inv;
    }
    __syncthreads();

    // ---------------- Phase 4: V = X @ W_v^T + b_v  (all-LDS K) -------------
#pragma unroll
    for (int i = 0; i < 10; ++i) {
        int tile = wave + 8 * i;
        int mt = tile >> 4, nt = tile & 15;
        v8f acc = (v8f)0.0f;
#pragma unroll
        for (int kt = 0; kt < WV_KT; ++kt) {
            v16bf a = load_a(sX, mt * 16, 264, kt * 32, m16, half);
            v16bf b = load_b(wvswz + ((kt * WV_NT + nt) * 32 + lane) * 8);
            acc = __builtin_amdgcn_wmma_f32_16x16x32_bf16(
                false, a, false, b, (short)0, acc, false, false);
        }
        int n = nt * 16 + m16;
        float bvv = b_v[n];
#pragma unroll
        for (int ii = 0; ii < 8; ++ii) {
            int m = mt * 16 + ii + 8 * half;
            sV[m * 260 + n] = acc[ii] + bvv;
        }
    }
    __syncthreads();

    // ---------------- Phase 5: attention combine y[t,a] ---------------------
    for (int idx = tid; idx < TOK_PER_WG * ADIM; idx += 256) {
        int t = idx >> 8, a = idx & 255, h = a >> 5;
        const float* w  = sAt + (t * NHEADS + h) * NMOD;
        const float* vb = sV + (t * NMOD) * 260 + a;
        float y = w[0] * vb[0] + w[1] * vb[260] + w[2] * vb[520] +
                  w[3] * vb[780] + w[4] * vb[1040];
        sY[t * 264 + a] = (unsigned short)f2bf1(y);
    }
    __syncthreads();

    // ---------------- Phase 6: O = Y @ W_out^T + b_out, zero-mask, store ----
#pragma unroll
    for (int i = 0; i < 6; ++i) {
        int nt = wave + 8 * i;  // 0..47
        v8f acc = (v8f)0.0f;
#pragma unroll
        for (int kt = 0; kt < WOUT_KT; ++kt) {
            v16bf a = load_a(sY, 0, 264, kt * 32, m16, half);
            v16bf b = load_b(woutswz + ((kt * WOUT_NT + nt) * 32 + lane) * 8);
            acc = __builtin_amdgcn_wmma_f32_16x16x32_bf16(
                false, a, false, b, (short)0, acc, false, false);
        }
        int n = nt * 16 + m16;
        float bo = b_out[n];
#pragma unroll
        for (int ii = 0; ii < 8; ++ii) {
            int t = ii + 8 * half;
            float val = sHas[t] ? (acc[ii] + bo) : 0.0f;
            out[(tkbase + t) * DIM + n] = val;
        }
    }
}

// ---------------------------------------------------------------------------
extern "C" void kernel_launch(void* const* d_in, const int* in_sizes, int n_in,
                              void* d_out, int out_size, void* d_ws, size_t ws_size,
                              hipStream_t stream) {
    const float* s2   = (const float*)d_in[0];
    const float* s1   = (const float*)d_in[1];
    const float* Win  = (const float*)d_in[2];
    const float* bin  = (const float*)d_in[3];
    const float* Wk   = (const float*)d_in[4];
    const float* bk   = (const float*)d_in[5];
    const float* Wv   = (const float*)d_in[6];
    const float* bv   = (const float*)d_in[7];
    const float* Wout = (const float*)d_in[8];
    const float* bout = (const float*)d_in[9];
    const float* qry  = (const float*)d_in[10];
    const int*   s2m  = (const int*)d_in[11];
    const int*   s1m  = (const int*)d_in[12];
    float* out = (float*)d_out;

    char* ws = (char*)d_ws;
    unsigned* winswz  = (unsigned*)(ws);                     // 393216 B
    unsigned* wvswz   = (unsigned*)(ws + 393216);            // 131072 B
    unsigned* woutswz = (unsigned*)(ws + 524288);            // 393216 B
    float*    qk      = (float*)(ws + 917504);               // 8192 B
    float*    qb      = (float*)(ws + 925696);               // 32 B

    prep_qk<<<1, 256, 0, stream>>>(qry, Wk, bk, qk, qb);
    prep_weights<<<896, 256, 0, stream>>>(Win, Wv, Wout, winswz, wvswz, woutswz);
    chnattn_main<<<BN / TOK_PER_WG, 256, SMEM_SZ, stream>>>(
        s2, s1, bin, bv, bout, s2m, s1m, winswz, wvswz, woutswz, qk, qb, out);
}
